// LlamaAttentionH2O_81174881894723
// MI455X (gfx1250) — matrix operand order repaired and checked by reading, not compiled, not run on hardware
//
#include <hip/hip_runtime.h>
#include <hip/hip_fp16.h>

// ---------------------------------------------------------------------------
// MI455X (gfx1250, wave32) Llama attention + H2O heavy-hitter mask.
// - All matmuls: v_wmma_f32_16x16x32_f16 (f16 operands, f32 accumulate).
// - LDS staging via Tensor Data Mover (tensor_load_to_lds, TENSORcnt),
//   double-buffered so DMA overlaps WMMA. Sync fallback if builtin absent.
// - Flash-style 2-pass attention: exact H2O column sums, no SxS tensor.
// ---------------------------------------------------------------------------

typedef __attribute__((ext_vector_type(16))) _Float16 v16h;
typedef __attribute__((ext_vector_type(8)))  float    v8f;
typedef __attribute__((ext_vector_type(4)))  unsigned int u32x4;
typedef __attribute__((ext_vector_type(8)))  int      i32x8;
typedef __attribute__((ext_vector_type(4)))  int      i32x4;

#if defined(__has_builtin)
#if __has_builtin(__builtin_amdgcn_tensor_load_to_lds) && __has_builtin(__builtin_amdgcn_s_wait_tensorcnt)
#define HAS_TDM 1
#endif
#endif
#ifndef HAS_TDM
#define HAS_TDM 0
#endif

// Toolchain arity probe: the gfx1250 TDM header ships only with the
// toolchain whose tensor_load_to_lds builtin takes 6 args (clang-23 /
// therock-10.0); ROCm 7.2 clang-22 has the 5-arg form.
#if defined(__has_include)
#if __has_include(<hip/amd_detail/amd_gfx1250_TDM.h>)
#define TDM_6ARG 1
#endif
#endif
#ifndef TDM_6ARG
#define TDM_6ARG 0
#endif

#define SEQ      2048
#define HID      4096
#define NH       32
#define NKV      8
#define HD       128
#define KVW      (NKV * HD)           // 1024
#define SCALE    0.08838834764831845f // 1/sqrt(128)
#define HEAVY    204
#define RECENT   204
#define SELN     (SEQ - RECENT)       // 1844
#define NEG_BIG  (-3.0e38f)
#define NEG_GRD  (-1.0e37f)

__device__ __forceinline__ v16h load16(const _Float16* p0, const _Float16* p1) {
    v16h r;
    uint4* u = (uint4*)&r;
    u[0] = *(const uint4*)p0;
    u[1] = *(const uint4*)p1;
    return r;
}
__device__ __forceinline__ v8f zero8() {
    v8f z = {0.f, 0.f, 0.f, 0.f, 0.f, 0.f, 0.f, 0.f};
    return z;
}
__device__ __forceinline__ v8f wmma16(v16h a, v16h b, v8f c) {
    return __builtin_amdgcn_wmma_f32_16x16x32_f16(false, a, false, b, (short)0, c, false, false);
}

#if HAS_TDM
// Byte offset of a __shared__ object within the workgroup LDS allocation.
__device__ __forceinline__ unsigned lds_off(const void* p) {
    return (unsigned)(unsigned long long)(const __attribute__((address_space(3))) void*)p;
}
// Issue one 2D TDM load: tile tileW x tileH f16 elements, row stride
// `strideElems`, from global `gp` into LDS byte offset `ldsByte`.
__device__ __forceinline__ void tdm_load_2d(unsigned ldsByte, const void* gp,
                                            unsigned tileW, unsigned tileH,
                                            unsigned long long strideElems) {
    unsigned long long ga = (unsigned long long)gp;
    const unsigned td0 = 0x40000000u, td1 = 0x40000000u; // huge dims: no OOB clip
    u32x4 g0;
    g0[0] = 1u;                                               // count=1, user mode
    g0[1] = ldsByte;                                          // lds_addr (bytes)
    g0[2] = (unsigned)ga;                                     // global_addr[31:0]
    g0[3] = (unsigned)((ga >> 32) & 0x1FFFFFFu) | (2u << 30); // addr[56:32] | type=2
    i32x8 g1;
    g1[0] = (int)(1u << 16);                                  // data_size=1 (2 bytes)
    g1[1] = (int)((td0 & 0xFFFFu) << 16);                     // tensor_dim0 lo
    g1[2] = (int)(((td0 >> 16) & 0xFFFFu) | ((td1 & 0xFFFFu) << 16));
    g1[3] = (int)(((td1 >> 16) & 0xFFFFu) | (tileW << 16));   // tile_dim0
    g1[4] = (int)(tileH & 0xFFFFu);                           // tile_dim1, tile_dim2=0
    g1[5] = (int)(strideElems & 0xFFFFFFFFu);                 // tensor_dim0_stride lo
    g1[6] = (int)((strideElems >> 32) & 0xFFFFu);             // stride hi
    g1[7] = 0;
    i32x4 z4 = {0, 0, 0, 0};
#if TDM_6ARG
    i32x8 z8 = {0, 0, 0, 0, 0, 0, 0, 0};
    __builtin_amdgcn_tensor_load_to_lds(g0, g1, z4, z4, z8, 0);
#else
    __builtin_amdgcn_tensor_load_to_lds(g0, g1, z4, z4, 0);
#endif
}
#endif

// ---------------------------------------------------------------------------
__global__ void cast_f32_f16(const float* __restrict__ in, _Float16* __restrict__ out, int n) {
    int i = blockIdx.x * blockDim.x + threadIdx.x;
    if (i < n) out[i] = (_Float16)in[i];
}

__global__ void zero_f32(float* __restrict__ p, int n) {
    int i = blockIdx.x * blockDim.x + threadIdx.x;
    if (i < n) p[i] = 0.f;
}

// RoPE on f32 GEMM output, write f16. ncols = 4096 (Q) or 1024 (K).
__global__ void rope_cast(const float* __restrict__ in, _Float16* __restrict__ out, int ncols) {
    int idx = blockIdx.x * blockDim.x + threadIdx.x;
    if (idx >= SEQ * ncols) return;
    int s   = idx / ncols;
    int col = idx - s * ncols;
    int d   = col & (HD - 1);
    int i2  = d & 63;
    float inv = expf(-(float)i2 * (9.210340371976184f / 64.0f)); // 10000^(-i/64)
    float fr  = (float)s * inv;
    float sn, cs;
    sincosf(fr, &sn, &cs);
    float x   = in[idx];
    float rot = (d < 64) ? -in[idx + 64] : in[idx - 64];
    out[idx] = (_Float16)(x * cs + rot * sn);
}

// ---------------------------------------------------------------------------
// C[M,N] (f32) = A[M,K] (f16 row-major) @ Bt[N,K]^T (f16 row-major).
// Block tile 256x128, 8 waves (4x2), wave tile 64x64, K step 32.
// TDM double-buffered LDS staging (fallback: cooperative sync staging).
__global__ __launch_bounds__(256) void gemm_f16_bt(
    const _Float16* __restrict__ A, const _Float16* __restrict__ Bt,
    float* __restrict__ C, int M, int N, int K) {
#if HAS_TDM
    __shared__ __align__(16) _Float16 aT[2][256 * 32];
    __shared__ __align__(16) _Float16 bT[2][128 * 32];
#else
    __shared__ __align__(16) _Float16 aT[1][256 * 32];
    __shared__ __align__(16) _Float16 bT[1][128 * 32];
#endif
    const int t = threadIdx.x, lane = t & 31, wave = t >> 5;
    const int hl = (lane >= 16), lc = lane & 15;
    const int m0 = blockIdx.y * 256;
    const int n0 = blockIdx.x * 128;
    const int wm = (wave >> 1) * 64;
    const int wn = (wave & 1) * 64;

    v8f acc[4][4];
#pragma unroll
    for (int i = 0; i < 4; ++i)
#pragma unroll
        for (int j = 0; j < 4; ++j) acc[i][j] = zero8();

#if HAS_TDM
    if (wave == 0) {
        tdm_load_2d(lds_off(&aT[0][0]), &A[(size_t)m0 * K], 32, 256, (unsigned long long)K);
        tdm_load_2d(lds_off(&bT[0][0]), &Bt[(size_t)n0 * K], 32, 128, (unsigned long long)K);
    }
#endif
    for (int k0 = 0; k0 < K; k0 += 32) {
#if HAS_TDM
        const int cur = (k0 >> 5) & 1;
        if (wave == 0) {
            if (k0 + 32 < K) {
                tdm_load_2d(lds_off(&aT[cur ^ 1][0]), &A[(size_t)m0 * K + k0 + 32],
                            32, 256, (unsigned long long)K);
                tdm_load_2d(lds_off(&bT[cur ^ 1][0]), &Bt[(size_t)n0 * K + k0 + 32],
                            32, 128, (unsigned long long)K);
                __builtin_amdgcn_s_wait_tensorcnt(2); // previous buffer landed
            } else {
                __builtin_amdgcn_s_wait_tensorcnt(0);
            }
        }
        __syncthreads();
#else
        const int cur = 0;
        __syncthreads();
#pragma unroll
        for (int c = 0; c < 4; ++c) {
            int chunk = t + c * 256;
            int r = chunk >> 2, k8 = (chunk & 3) * 8;
            *(uint4*)&aT[0][r * 32 + k8] = *(const uint4*)&A[(size_t)(m0 + r) * K + k0 + k8];
        }
#pragma unroll
        for (int c = 0; c < 2; ++c) {
            int chunk = t + c * 256;
            int r = chunk >> 2, k8 = (chunk & 3) * 8;
            *(uint4*)&bT[0][r * 32 + k8] = *(const uint4*)&Bt[(size_t)(n0 + r) * K + k0 + k8];
        }
        if (k0 + 32 < K) {
            int r = t >> 1;
            __builtin_prefetch(&A[(size_t)(m0 + r) * K + k0 + 32], 0, 0);
            __builtin_prefetch(&Bt[(size_t)(n0 + (r & 127)) * K + k0 + 32], 0, 0);
        }
        __syncthreads();
#endif
        const int kbA = hl ? 8 : 0, kbB = hl ? 8 : 0;
        v16h bf[4], af[4];
#pragma unroll
        for (int sn = 0; sn < 4; ++sn) {
            const _Float16* p = &bT[cur][(wn + sn * 16 + lc) * 32 + kbB];
            bf[sn] = load16(p, p + 16);
        }
#pragma unroll
        for (int sm = 0; sm < 4; ++sm) {
            const _Float16* p = &aT[cur][(wm + sm * 16 + lc) * 32 + kbA];
            af[sm] = load16(p, p + 16);
        }
#pragma unroll
        for (int sm = 0; sm < 4; ++sm)
#pragma unroll
            for (int sn = 0; sn < 4; ++sn)
                acc[sm][sn] = wmma16(af[sm], bf[sn], acc[sm][sn]);
        __syncthreads();
    }
#pragma unroll
    for (int sm = 0; sm < 4; ++sm)
#pragma unroll
        for (int sn = 0; sn < 4; ++sn) {
            int col = n0 + wn + sn * 16 + lc;
#pragma unroll
            for (int i = 0; i < 8; ++i) {
                int row = m0 + wm + sm * 16 + (hl ? 8 + i : i);
                C[(size_t)row * N + col] = acc[sm][sn][i];
            }
        }
}

// ---------------------------------------------------------------------------
// Pass 1: per-row softmax max (M) and denominator (L).
// Grid: (SEQ/128, NH). Block 256 = 8 waves; wave w owns q rows w*16..w*16+15.
__global__ __launch_bounds__(256) void attn_pass1(
    const _Float16* __restrict__ Qh, const _Float16* __restrict__ Kh,
    float* __restrict__ Mbuf, float* __restrict__ Lbuf) {
#if HAS_TDM
    __shared__ __align__(16) _Float16 kT[2][128 * 128];
#else
    __shared__ __align__(16) _Float16 kT[1][128 * 128];
#endif
    const int t = threadIdx.x, lane = t & 31, w = t >> 5;
    const int hl = (lane >= 16), lc = lane & 15;
    const int qb = blockIdx.x, h = blockIdx.y, hkv = h >> 2;
    const int qrow = qb * 128 + w * 16 + lc;

    v16h qa[4];
    const _Float16* qbase = Qh + (size_t)qrow * HID + h * HD;
#pragma unroll
    for (int kk = 0; kk < 4; ++kk) {
        const _Float16* p = qbase + kk * 32 + (hl ? 8 : 0);
        qa[kk] = load16(p, p + 16);
    }
    float m[8], l[8];
#pragma unroll
    for (int i = 0; i < 8; ++i) { m[i] = NEG_BIG; l[i] = 0.f; }

#if HAS_TDM
    if (w == 0)
        tdm_load_2d(lds_off(&kT[0][0]), &Kh[(size_t)0 * KVW + hkv * HD], 128, 128,
                    (unsigned long long)KVW);
#endif
    for (int kt = 0; kt <= qb; ++kt) {
#if HAS_TDM
        const int cur = kt & 1;
        if (w == 0) {
            if (kt < qb) {
                tdm_load_2d(lds_off(&kT[cur ^ 1][0]),
                            &Kh[(size_t)((kt + 1) * 128) * KVW + hkv * HD], 128, 128,
                            (unsigned long long)KVW);
                __builtin_amdgcn_s_wait_tensorcnt(1);
            } else {
                __builtin_amdgcn_s_wait_tensorcnt(0);
            }
        }
        __syncthreads();
#else
        const int cur = 0;
        __syncthreads();
#pragma unroll
        for (int c = 0; c < 8; ++c) {
            int chunk = t + c * 256;
            int r = chunk >> 4, d8 = (chunk & 15) * 8;
            *(uint4*)&kT[0][r * 128 + d8] =
                *(const uint4*)&Kh[(size_t)(kt * 128 + r) * KVW + hkv * HD + d8];
        }
        __syncthreads();
#endif
        for (int nt = 0; nt < 8; ++nt) {
            v8f c8 = zero8();
#pragma unroll
            for (int kk = 0; kk < 4; ++kk) {
                const _Float16* p = &kT[cur][(nt * 16 + lc) * 128 + kk * 32 + (hl ? 8 : 0)];
                c8 = wmma16(qa[kk], load16(p, p + 16), c8);
            }
            int kcol = kt * 128 + nt * 16 + lc;
#pragma unroll
            for (int i = 0; i < 8; ++i) {
                int qg  = qb * 128 + w * 16 + (hl ? 8 + i : i);
                float v = (kcol <= qg) ? c8[i] * SCALE : NEG_BIG;
                float mx = v;
#pragma unroll
                for (int off = 1; off < 16; off <<= 1) mx = fmaxf(mx, __shfl_xor(mx, off, 16));
                float nm = fmaxf(m[i], mx);
                if (nm > NEG_GRD) {
                    float e = __expf(v - nm);
#pragma unroll
                    for (int off = 1; off < 16; off <<= 1) e += __shfl_xor(e, off, 16);
                    l[i] = l[i] * __expf(m[i] - nm) + e;
                    m[i] = nm;
                }
            }
        }
        __syncthreads();
    }
    if (lc == 0) {
#pragma unroll
        for (int i = 0; i < 8; ++i) {
            int row = qb * 128 + w * 16 + (hl ? 8 + i : i);
            Mbuf[(size_t)h * SEQ + row] = m[i];
            Lbuf[(size_t)h * SEQ + row] = l[i];
        }
    }
}

// ---------------------------------------------------------------------------
// Pass 2: exact probabilities -> H2O column sums + out = P @ V (WMMA).
__global__ __launch_bounds__(256) void attn_pass2(
    const _Float16* __restrict__ Qh, const _Float16* __restrict__ Kh,
    const _Float16* __restrict__ Vh, const float* __restrict__ Mbuf,
    const float* __restrict__ Lbuf, float* __restrict__ colsum,
    _Float16* __restrict__ Obuf) {
    __shared__ __align__(16) _Float16 kT[128 * 128];
    __shared__ __align__(16) _Float16 vT[128 * 128]; // transposed: [d][k]
    __shared__ __align__(16) _Float16 pT[128 * 128]; // probabilities [q][k]
    __shared__ float csL[128];
    const int t = threadIdx.x, lane = t & 31, w = t >> 5;
    const int hl = (lane >= 16), lc = lane & 15;
    const int qb = blockIdx.x, h = blockIdx.y, hkv = h >> 2;
    const int qrow = qb * 128 + w * 16 + lc;

    v16h qa[4];
    const _Float16* qbase = Qh + (size_t)qrow * HID + h * HD;
#pragma unroll
    for (int kk = 0; kk < 4; ++kk) {
        const _Float16* p = qbase + kk * 32 + (hl ? 8 : 0);
        qa[kk] = load16(p, p + 16);
    }
    float mr[8], ir[8];
#pragma unroll
    for (int i = 0; i < 8; ++i) {
        int qg = qb * 128 + w * 16 + (hl ? 8 + i : i);
        mr[i] = Mbuf[(size_t)h * SEQ + qg];
        float lv = Lbuf[(size_t)h * SEQ + qg];
        ir[i] = (lv > 0.f) ? 1.f / lv : 0.f;
    }
    v8f oacc[8];
#pragma unroll
    for (int i = 0; i < 8; ++i) oacc[i] = zero8();

    for (int kt = 0; kt <= qb; ++kt) {
        __syncthreads();
        if (t < 128) csL[t] = 0.f;
#if HAS_TDM
        if (w == 0)
            tdm_load_2d(lds_off(&kT[0]), &Kh[(size_t)(kt * 128) * KVW + hkv * HD], 128, 128,
                        (unsigned long long)KVW);
#endif
#pragma unroll
        for (int c = 0; c < 8; ++c) {
            int chunk = t + c * 256;
            int r = chunk >> 4, d8 = (chunk & 15) * 8;
#if !HAS_TDM
            *(uint4*)&kT[r * 128 + d8] =
                *(const uint4*)&Kh[(size_t)(kt * 128 + r) * KVW + hkv * HD + d8];
#endif
            uint4 raw = *(const uint4*)&Vh[(size_t)(kt * 128 + r) * KVW + hkv * HD + d8];
            _Float16 tmp[8];
            *(uint4*)tmp = raw;
#pragma unroll
            for (int j = 0; j < 8; ++j) vT[(d8 + j) * 128 + r] = tmp[j];
        }
#if HAS_TDM
        if (w == 0) __builtin_amdgcn_s_wait_tensorcnt(0);
#endif
        __syncthreads();
        // scores -> probabilities
        for (int nt = 0; nt < 8; ++nt) {
            v8f c8 = zero8();
#pragma unroll
            for (int kk = 0; kk < 4; ++kk) {
                const _Float16* p = &kT[(nt * 16 + lc) * 128 + kk * 32 + (hl ? 8 : 0)];
                c8 = wmma16(qa[kk], load16(p, p + 16), c8);
            }
            int kcol = kt * 128 + nt * 16 + lc;
            float part = 0.f;
#pragma unroll
            for (int i = 0; i < 8; ++i) {
                int qg  = qb * 128 + w * 16 + (hl ? 8 + i : i);
                float pv = (kcol <= qg) ? __expf(c8[i] * SCALE - mr[i]) * ir[i] : 0.f;
                part += pv;
                pT[(w * 16 + (hl ? 8 + i : i)) * 128 + nt * 16 + lc] = (_Float16)pv;
            }
            atomicAdd(&csL[nt * 16 + lc], part);
        }
        __syncthreads();
        // out += P @ V
#pragma unroll
        for (int kk = 0; kk < 4; ++kk) {
            const _Float16* pp = &pT[(w * 16 + lc) * 128 + kk * 32 + (hl ? 8 : 0)];
            v16h pa = load16(pp, pp + 16);
#pragma unroll
            for (int dt = 0; dt < 8; ++dt) {
                const _Float16* vp = &vT[(dt * 16 + lc) * 128 + kk * 32 + (hl ? 8 : 0)];
                oacc[dt] = wmma16(pa, load16(vp, vp + 16), oacc[dt]);
            }
        }
        if (t < 128) atomicAdd(&colsum[(size_t)h * SEQ + kt * 128 + t], csL[t]);
    }
#pragma unroll
    for (int dt = 0; dt < 8; ++dt) {
        int dcol = dt * 16 + lc;
#pragma unroll
        for (int i = 0; i < 8; ++i) {
            int row = qb * 128 + w * 16 + (hl ? 8 + i : i);
            Obuf[(size_t)row * HID + h * HD + dcol] = (_Float16)oacc[dt][i];
        }
    }
}

// ---------------------------------------------------------------------------
// H2O top-k + mask emission. One block per head.
__global__ __launch_bounds__(256) void topk_mask(const float* __restrict__ colsum,
                                                 float* __restrict__ mask) {
    __shared__ float vals[SELN];
    __shared__ float rv[256];
    __shared__ int   rix[256];
    const int h = blockIdx.x, t = threadIdx.x;
    for (int j = t; j < SELN; j += 256) vals[j] = colsum[(size_t)h * SEQ + j];
    for (int j = t; j <= SEQ; j += 256)
        mask[(size_t)h * (SEQ + 1) + j] = (j >= SEQ + 1 - RECENT) ? 1.f : 0.f;
    __syncthreads();
    for (int it = 0; it < HEAVY; ++it) {
        float bv = -3.4e38f; int bi = -1;
        for (int j = t; j < SELN; j += 256) {
            float v = vals[j];
            if (v > bv) { bv = v; bi = j; }
        }
        rv[t] = bv; rix[t] = bi;
        __syncthreads();
        for (int s2 = 128; s2 > 0; s2 >>= 1) {
            if (t < s2) {
                float ov = rv[t + s2]; int oi = rix[t + s2];
                if (ov > rv[t] || (ov == rv[t] && oi >= 0 && oi < rix[t])) { rv[t] = ov; rix[t] = oi; }
            }
            __syncthreads();
        }
        if (t == 0) {
            mask[(size_t)h * (SEQ + 1) + rix[0]] = 1.f;
            vals[rix[0]] = -3.4e38f;
        }
        __syncthreads();
    }
}

// ---------------------------------------------------------------------------
extern "C" void kernel_launch(void* const* d_in, const int* in_sizes, int n_in,
                              void* d_out, int out_size, void* d_ws, size_t ws_size,
                              hipStream_t stream) {
    const float* hs = (const float*)d_in[0];
    const float* Wq = (const float*)d_in[1];
    const float* Wk = (const float*)d_in[2];
    const float* Wv = (const float*)d_in[3];
    const float* Wo = (const float*)d_in[4];
    float* out = (float*)d_out;                 // [SEQ*HID] attn_output
    float* maskOut = out + (size_t)SEQ * HID;   // [NH*(SEQ+1)] attention_masks_next

    char* ws = (char*)d_ws;
    size_t off = 0;
    auto take = [&](size_t bytes) -> char* {
        char* p = ws + off;
        off += (bytes + 255) & ~(size_t)255;
        return p;
    };
    _Float16* hsH = (_Float16*)take((size_t)SEQ * HID * 2);
    _Float16* WqH = (_Float16*)take((size_t)HID * HID * 2);
    _Float16* WkH = (_Float16*)take((size_t)KVW * HID * 2);
    _Float16* WvH = (_Float16*)take((size_t)KVW * HID * 2);
    _Float16* WoH = (_Float16*)take((size_t)HID * HID * 2);
    float*    qF  = (float*)take((size_t)SEQ * HID * 4);
    float*    kF  = (float*)take((size_t)SEQ * KVW * 4);
    float*    vF  = (float*)take((size_t)SEQ * KVW * 4);
    _Float16* Qf  = (_Float16*)take((size_t)SEQ * HID * 2);
    _Float16* Kf  = (_Float16*)take((size_t)SEQ * KVW * 2);
    _Float16* Vf  = (_Float16*)take((size_t)SEQ * KVW * 2);
    float*    Mb  = (float*)take((size_t)NH * SEQ * 4);
    float*    Lb  = (float*)take((size_t)NH * SEQ * 4);
    float*    cs  = (float*)take((size_t)NH * SEQ * 4);
    _Float16* Ob  = (_Float16*)take((size_t)SEQ * HID * 2);

    auto blocks = [](size_t n) { return (unsigned)((n + 255) / 256); };

    // 1) f32 -> f16 casts
    cast_f32_f16<<<blocks((size_t)SEQ * HID), 256, 0, stream>>>(hs, hsH, SEQ * HID);
    cast_f32_f16<<<blocks((size_t)HID * HID), 256, 0, stream>>>(Wq, WqH, HID * HID);
    cast_f32_f16<<<blocks((size_t)KVW * HID), 256, 0, stream>>>(Wk, WkH, KVW * HID);
    cast_f32_f16<<<blocks((size_t)KVW * HID), 256, 0, stream>>>(Wv, WvH, KVW * HID);
    cast_f32_f16<<<blocks((size_t)HID * HID), 256, 0, stream>>>(Wo, WoH, HID * HID);

    // 2) Q/K/V projections (A @ W^T via Bt layout); block tile 256x128
    gemm_f16_bt<<<dim3(HID / 128, SEQ / 256), 256, 0, stream>>>(hsH, WqH, qF, SEQ, HID, HID);
    gemm_f16_bt<<<dim3(KVW / 128, SEQ / 256), 256, 0, stream>>>(hsH, WkH, kF, SEQ, KVW, HID);
    gemm_f16_bt<<<dim3(KVW / 128, SEQ / 256), 256, 0, stream>>>(hsH, WvH, vF, SEQ, KVW, HID);

    // 3) RoPE (Q, K) + cast, V cast
    rope_cast<<<blocks((size_t)SEQ * HID), 256, 0, stream>>>(qF, Qf, HID);
    rope_cast<<<blocks((size_t)SEQ * KVW), 256, 0, stream>>>(kF, Kf, KVW);
    cast_f32_f16<<<blocks((size_t)SEQ * KVW), 256, 0, stream>>>(vF, Vf, SEQ * KVW);

    // 4) attention (2-pass flash with exact H2O column sums)
    zero_f32<<<blocks((size_t)NH * SEQ), 256, 0, stream>>>(cs, NH * SEQ);
    attn_pass1<<<dim3(SEQ / 128, NH), 256, 0, stream>>>(Qf, Kf, Mb, Lb);
    attn_pass2<<<dim3(SEQ / 128, NH), 256, 0, stream>>>(Qf, Kf, Vf, Mb, Lb, cs, Ob);

    // 5) output projection -> d_out
    gemm_f16_bt<<<dim3(HID / 128, SEQ / 256), 256, 0, stream>>>(Ob, WoH, out, SEQ, HID, HID);

    // 6) H2O mask
    topk_mask<<<NH, 256, 0, stream>>>(cs, maskOut);
}
